// GCNBaseline_61194694033409
// MI455X (gfx1250) — matrix operand order repaired and checked by reading, not compiled, not run on hardware
//
#include <hip/hip_runtime.h>
#include <hip/hip_bf16.h>

// MI455X / gfx1250, wave32. bf16 WMMA (16x16x32) with f32 accumulation.

typedef __attribute__((ext_vector_type(16))) __bf16 v16bf;
typedef __attribute__((ext_vector_type(8)))  __bf16 v8bf;
typedef __attribute__((ext_vector_type(8)))  float  v8f;

#define BDIM 512
#define NDIM 400
#define DDIM 128
#define EPS_BN 1e-5f
#define EPS_LN 1e-5f

// ---------------------------------------------------------------------------
// dinv[b,n] = rsqrt(1 + sum_m sc[b,n,m])   (diag of sc is zero; self-loop = 1)
__global__ void k_dinv(const float* __restrict__ sc, float* __restrict__ dinv) {
    int wave = blockIdx.x * (blockDim.x >> 5) + (threadIdx.x >> 5);
    int lane = threadIdx.x & 31;
    if (wave >= BDIM * NDIM) return;
    const float* row = sc + (size_t)wave * NDIM;
    float s = 0.f;
    for (int m = lane; m < NDIM; m += 32) s += row[m];
    #pragma unroll
    for (int off = 16; off > 0; off >>= 1) s += __shfl_down(s, off, 32);
    if (lane == 0) dinv[wave] = rsqrtf(s + 1.0f);
}

// a_bf16[b,n,m] = bf16( (sc + I) * dinv[n] * dinv[m] )
__global__ void k_build_a(const float* __restrict__ sc, const float* __restrict__ dinv,
                          __bf16* __restrict__ a) {
    size_t total = (size_t)BDIM * NDIM * NDIM;
    for (size_t idx = (size_t)blockIdx.x * blockDim.x + threadIdx.x; idx < total;
         idx += (size_t)gridDim.x * blockDim.x) {
        size_t bn = idx / NDIM;
        int m = (int)(idx - bn * NDIM);
        size_t b = bn / NDIM;
        int n = (int)(bn - b * NDIM);
        float v = sc[idx] + ((n == m) ? 1.0f : 0.0f);
        v *= dinv[bn] * dinv[b * NDIM + m];
        a[idx] = (__bf16)v;
    }
}

__global__ void k_cvt_bf16(const float* __restrict__ in, __bf16* __restrict__ out, size_t n) {
    for (size_t i = (size_t)blockIdx.x * blockDim.x + threadIdx.x; i < n;
         i += (size_t)gridDim.x * blockDim.x)
        out[i] = (__bf16)in[i];
}

// ---------------------------------------------------------------------------
// WMMA GEMM core helpers (128x128 block tile, 256 threads = 8 waves).
// LDS layout: As[buf][m][k] (row-major), Bs[buf][n][k] (B columns K-contiguous)
// A-fragment layout (16-bit A 16x32): lane-half0 -> K 0..7 & 16..23,
// half1 -> 8..15 & 24..31, row = lane&15.
// B-fragment: half0 -> K 0..7 & 16..23, half1 -> K 8..15 & 24..31, col = lane&15.
// C/D: VGPR r -> M=r (half0) / M=8+r (half1), N = lane&15.

// Generic flat GEMM: C[MxN] = A[MxK] @ B[KxN]
//  - A bf16 (or f32 converted on the fly), row-major, lda = K
//  - B bf16 row-major [K x N]
//  - OUT_BF16 ? bf16 store : f32 store (+bias)
// Assumes: M % 128 == 0, N % 128 == 0, K % 16 == 0.
template <bool A_IS_F32, bool OUT_BF16>
__global__ void __launch_bounds__(256) k_gemm(const void* __restrict__ Av,
                                              const __bf16* __restrict__ B,
                                              void* __restrict__ Cv,
                                              const float* __restrict__ bias,
                                              int M, int N, int K) {
    __shared__ __bf16 As[2][128][40];
    __shared__ __bf16 Bs[2][128][40];
    const int tid = threadIdx.x;
    const int wave = tid >> 5, lane = tid & 31;
    const int lhalf = lane >> 4, l15 = lane & 15;
    const int m0 = blockIdx.x * 128, n0 = blockIdx.y * 128;

    const int arow = tid >> 1, akk = (tid & 1) * 16;   // A: 16 elems along K
    const int bk = tid >> 3, bnn = (tid & 7) * 16;     // B: 16 elems along N

    v8bf zz;
    #pragma unroll
    for (int i = 0; i < 8; i++) zz[i] = (__bf16)0.f;

    v8f acc[8];
    #pragma unroll
    for (int t = 0; t < 8; t++)
        #pragma unroll
        for (int r = 0; r < 8; r++) acc[t][r] = 0.f;

    const int nchunks = (K + 31) >> 5;
    v8bf sa0 = zz, sa1 = zz, sb0 = zz, sb1 = zz;

    // fetch chunk c from global into staging registers (validity at 16-granularity)
    auto fetch = [&](int c) {
        const int k0 = c << 5;
        if (k0 + akk + 16 <= K) {
            if (A_IS_F32) {
                const float* s = (const float*)Av + (size_t)(m0 + arow) * K + k0 + akk;
                #pragma unroll
                for (int i = 0; i < 8; i++) { sa0[i] = (__bf16)s[i]; sa1[i] = (__bf16)s[8 + i]; }
            } else {
                const __bf16* s = (const __bf16*)Av + (size_t)(m0 + arow) * K + k0 + akk;
                sa0 = *(const v8bf*)s; sa1 = *(const v8bf*)(s + 8);
            }
        } else { sa0 = zz; sa1 = zz; }
        if (k0 + bk < K) {
            const __bf16* s = B + (size_t)(k0 + bk) * N + n0 + bnn;
            sb0 = *(const v8bf*)s; sb1 = *(const v8bf*)(s + 8);
        } else { sb0 = zz; sb1 = zz; }
    };
    // commit staged registers into LDS buffer (c & 1)
    auto commit = [&](int c) {
        const int buf = c & 1;
        #pragma unroll
        for (int i = 0; i < 8; i++) { As[buf][arow][akk + i] = sa0[i]; As[buf][arow][akk + 8 + i] = sa1[i]; }
        #pragma unroll
        for (int i = 0; i < 8; i++) { Bs[buf][bnn + i][bk] = sb0[i]; Bs[buf][bnn + 8 + i][bk] = sb1[i]; }
    };

    fetch(0); commit(0);
    __syncthreads();

    for (int c = 0; c < nchunks; c++) {
        const int buf = c & 1;
        if (c + 1 < nchunks) fetch(c + 1);       // next-tile global loads issue early

        v16bf af;
        {
            const __bf16* p = &As[buf][wave * 16 + l15][lhalf * 8];
            v8bf lo = *(const v8bf*)p;
            v8bf hi = *(const v8bf*)(p + 16);
            #pragma unroll
            for (int i = 0; i < 8; i++) { af[i] = lo[i]; af[8 + i] = hi[i]; }
        }
        v16bf bfr[8];                             // distinct regs -> one dscnt wait
        #pragma unroll
        for (int t = 0; t < 8; t++) {
            const __bf16* q = &Bs[buf][t * 16 + l15][lhalf * 8];
            v8bf b0 = *(const v8bf*)q;
            v8bf b1 = *(const v8bf*)(q + 16);
            #pragma unroll
            for (int i = 0; i < 8; i++) { bfr[t][i] = b0[i]; bfr[t][8 + i] = b1[i]; }
        }
        #pragma unroll
        for (int t = 0; t < 8; t++)
            acc[t] = __builtin_amdgcn_wmma_f32_16x16x32_bf16(
                false, af, false, bfr[t], (short)0, acc[t], false, false);

        if (c + 1 < nchunks) commit(c + 1);      // stores into the other buffer
        __syncthreads();
    }

    #pragma unroll
    for (int t = 0; t < 8; t++) {
        int col = n0 + t * 16 + l15;
        #pragma unroll
        for (int r = 0; r < 8; r++) {
            int row = m0 + wave * 16 + lhalf * 8 + r;
            float v = acc[t][r];
            if (OUT_BF16) {
                ((__bf16*)Cv)[(size_t)row * N + col] = (__bf16)v;
            } else {
                ((float*)Cv)[(size_t)row * N + col] = v + (bias ? bias[col] : 0.f);
            }
        }
    }
}

// ---------------------------------------------------------------------------
// Batched aggregation: x_out[b] = BN( a[b](400x400) @ y[b](400x128) + bias ) (+ReLU)
// grid = (B, 4 m-tiles). Row edge handled by clamping loads (garbage rows are
// computed but never stored), K edge at 16-granularity zero-fill.
__global__ void __launch_bounds__(256) k_agg(const __bf16* __restrict__ a,
                                             const __bf16* __restrict__ y,
                                             __bf16* __restrict__ xout,
                                             const float* __restrict__ cb,
                                             const float* __restrict__ bn_g,
                                             const float* __restrict__ bn_b,
                                             const float* __restrict__ bn_m,
                                             const float* __restrict__ bn_v,
                                             int do_relu) {
    __shared__ __bf16 As[2][128][40];
    __shared__ __bf16 Bs[2][128][40];
    const int tid = threadIdx.x;
    const int wave = tid >> 5, lane = tid & 31;
    const int lhalf = lane >> 4, l15 = lane & 15;
    const int b = blockIdx.x, m0 = blockIdx.y * 128;
    const __bf16* A = a + (size_t)b * NDIM * NDIM;
    const __bf16* Y = y + (size_t)b * NDIM * DDIM;
    __bf16* X = xout + (size_t)b * NDIM * DDIM;

    const int arow = tid >> 1, akk = (tid & 1) * 16;
    const int bk = tid >> 3, bnn = (tid & 7) * 16;
    const int arow_g = (m0 + arow < NDIM) ? (m0 + arow) : (NDIM - 1);  // clamp

    v8bf zz;
    #pragma unroll
    for (int i = 0; i < 8; i++) zz[i] = (__bf16)0.f;

    v8f acc[8];
    #pragma unroll
    for (int t = 0; t < 8; t++)
        #pragma unroll
        for (int r = 0; r < 8; r++) acc[t][r] = 0.f;

    const int nchunks = (NDIM + 31) >> 5;   // 13, last has 16 valid K
    v8bf sa0 = zz, sa1 = zz, sb0 = zz, sb1 = zz;

    auto fetch = [&](int c) {
        const int k0 = c << 5;
        if (k0 + akk + 16 <= NDIM) {
            const __bf16* s = A + (size_t)arow_g * NDIM + k0 + akk;
            sa0 = *(const v8bf*)s; sa1 = *(const v8bf*)(s + 8);
        } else { sa0 = zz; sa1 = zz; }
        if (k0 + bk < NDIM) {
            const __bf16* s = Y + (size_t)(k0 + bk) * DDIM + bnn;
            sb0 = *(const v8bf*)s; sb1 = *(const v8bf*)(s + 8);
        } else { sb0 = zz; sb1 = zz; }
    };
    auto commit = [&](int c) {
        const int buf = c & 1;
        #pragma unroll
        for (int i = 0; i < 8; i++) { As[buf][arow][akk + i] = sa0[i]; As[buf][arow][akk + 8 + i] = sa1[i]; }
        #pragma unroll
        for (int i = 0; i < 8; i++) { Bs[buf][bnn + i][bk] = sb0[i]; Bs[buf][bnn + 8 + i][bk] = sb1[i]; }
    };

    fetch(0); commit(0);
    __syncthreads();

    for (int c = 0; c < nchunks; c++) {
        const int buf = c & 1;
        if (c + 1 < nchunks) fetch(c + 1);

        v16bf af;
        {
            const __bf16* p = &As[buf][wave * 16 + l15][lhalf * 8];
            v8bf lo = *(const v8bf*)p;
            v8bf hi = *(const v8bf*)(p + 16);
            #pragma unroll
            for (int i = 0; i < 8; i++) { af[i] = lo[i]; af[8 + i] = hi[i]; }
        }
        v16bf bfr[8];
        #pragma unroll
        for (int t = 0; t < 8; t++) {
            const __bf16* q = &Bs[buf][t * 16 + l15][lhalf * 8];
            v8bf b0 = *(const v8bf*)q;
            v8bf b1 = *(const v8bf*)(q + 16);
            #pragma unroll
            for (int i = 0; i < 8; i++) { bfr[t][i] = b0[i]; bfr[t][8 + i] = b1[i]; }
        }
        #pragma unroll
        for (int t = 0; t < 8; t++)
            acc[t] = __builtin_amdgcn_wmma_f32_16x16x32_bf16(
                false, af, false, bfr[t], (short)0, acc[t], false, false);

        if (c + 1 < nchunks) commit(c + 1);
        __syncthreads();
    }

    // fused bias + BN (+ReLU) epilogue, bf16 out
    #pragma unroll
    for (int t = 0; t < 8; t++) {
        int d = t * 16 + l15;
        float scale = bn_g[d] * rsqrtf(bn_v[d] + EPS_BN);
        float shift = bn_b[d] - bn_m[d] * scale;
        float bs = cb[d];
        #pragma unroll
        for (int r = 0; r < 8; r++) {
            int row = m0 + wave * 16 + lhalf * 8 + r;
            if (row < NDIM) {
                float v = (acc[t][r] + bs) * scale + shift;
                if (do_relu) v = fmaxf(v, 0.f);
                X[(size_t)row * DDIM + d] = (__bf16)v;
            }
        }
    }
}

// ---------------------------------------------------------------------------
// LayerNorm(+ReLU) over rows of length L, one wave per row.
__global__ void k_ln_relu(const float* __restrict__ in, float* __restrict__ out,
                          const float* __restrict__ g, const float* __restrict__ b,
                          int L, int rows, int do_relu) {
    int wave = blockIdx.x * (blockDim.x >> 5) + (threadIdx.x >> 5);
    int lane = threadIdx.x & 31;
    if (wave >= rows) return;
    const float* x = in + (size_t)wave * L;
    float s = 0.f, s2 = 0.f;
    for (int i = lane; i < L; i += 32) { float v = x[i]; s += v; s2 += v * v; }
    #pragma unroll
    for (int off = 16; off > 0; off >>= 1) { s += __shfl_down(s, off, 32); s2 += __shfl_down(s2, off, 32); }
    s = __shfl(s, 0, 32); s2 = __shfl(s2, 0, 32);
    float m = s / (float)L;
    float var = s2 / (float)L - m * m;
    float inv = rsqrtf(var + EPS_LN);
    float* o = out + (size_t)wave * L;
    for (int i = lane; i < L; i += 32) {
        float v = (x[i] - m) * inv * g[i] + b[i];
        if (do_relu) v = fmaxf(v, 0.f);
        o[i] = v;
    }
}

// Small dense tail GEMM, one thread per output element (f32).
__global__ void k_fc(const float* __restrict__ h, const float* __restrict__ w,
                     const float* __restrict__ bias, float* __restrict__ out,
                     int M, int N, int K) {
    int t = blockIdx.x * blockDim.x + threadIdx.x;
    if (t >= M * N) return;
    int r = t / N, c = t - r * N;
    float s = bias[c];
    for (int k = 0; k < K; k++) s += h[(size_t)r * K + k] * w[(size_t)k * N + c];
    out[t] = s;
}

// ---------------------------------------------------------------------------
extern "C" void kernel_launch(void* const* d_in, const int* in_sizes, int n_in,
                              void* d_out, int out_size, void* d_ws, size_t ws_size,
                              hipStream_t stream) {
    (void)in_sizes; (void)n_in; (void)out_size; (void)ws_size;
    // Input order (setup_inputs dict order, lists expanded):
    const float* sc = (const float*)d_in[1];                 // [512,400,400]
    const float* conv_w[3] = { (const float*)d_in[2], (const float*)d_in[3], (const float*)d_in[4] };
    const float* conv_b[3] = { (const float*)d_in[5], (const float*)d_in[6], (const float*)d_in[7] };
    const float* bn_g[3]   = { (const float*)d_in[8], (const float*)d_in[9], (const float*)d_in[10] };
    const float* bn_b[3]   = { (const float*)d_in[11], (const float*)d_in[12], (const float*)d_in[13] };
    const float* bn_m[3]   = { (const float*)d_in[14], (const float*)d_in[15], (const float*)d_in[16] };
    const float* bn_v[3]   = { (const float*)d_in[17], (const float*)d_in[18], (const float*)d_in[19] };
    const float* w1 = (const float*)d_in[20]; const float* b1 = (const float*)d_in[21];
    const float* ln1_g = (const float*)d_in[22]; const float* ln1_b = (const float*)d_in[23];
    const float* w2 = (const float*)d_in[24]; const float* b2 = (const float*)d_in[25];
    const float* ln2_g = (const float*)d_in[26]; const float* ln2_b = (const float*)d_in[27];
    const float* w3 = (const float*)d_in[28]; const float* b3 = (const float*)d_in[29];

    // workspace carve-out
    char* ws = (char*)d_ws;
    size_t off = 0;
    auto carve = [&](size_t bytes) { char* p = ws + off; off = (off + bytes + 255) & ~(size_t)255; return p; };
    float*  dinv  = (float*)carve((size_t)BDIM * NDIM * 4);
    __bf16* a_bf  = (__bf16*)carve((size_t)BDIM * NDIM * NDIM * 2);
    __bf16* y_bf  = (__bf16*)carve((size_t)BDIM * NDIM * DDIM * 2);
    __bf16* xA    = (__bf16*)carve((size_t)BDIM * NDIM * DDIM * 2);
    __bf16* xB    = (__bf16*)carve((size_t)BDIM * NDIM * DDIM * 2);
    __bf16* w0b   = (__bf16*)carve((size_t)NDIM * DDIM * 2);
    __bf16* w1b   = (__bf16*)carve((size_t)DDIM * DDIM * 2);
    __bf16* w2b   = (__bf16*)carve((size_t)DDIM * DDIM * 2);
    __bf16* wh1b  = (__bf16*)carve((size_t)NDIM * DDIM * 256 * 2);
    float*  h1raw = (float*)carve((size_t)BDIM * 256 * 4);
    float*  h1    = (float*)carve((size_t)BDIM * 256 * 4);
    float*  h2raw = (float*)carve((size_t)BDIM * 64 * 4);
    float*  h2    = (float*)carve((size_t)BDIM * 64 * 4);

    const int MROWS = BDIM * NDIM;  // 204800

    // 1) degree -> dinv
    k_dinv<<<(MROWS + 7) / 8, 256, 0, stream>>>(sc, dinv);
    // 2) normalized adjacency in bf16
    {
        size_t total = (size_t)BDIM * NDIM * NDIM;
        int blocks = (int)((total + 255) / 256);
        k_build_a<<<blocks, 256, 0, stream>>>(sc, dinv, a_bf);
    }
    // 3) weight conversions
    k_cvt_bf16<<<((size_t)NDIM * DDIM + 255) / 256, 256, 0, stream>>>(conv_w[0], w0b, (size_t)NDIM * DDIM);
    k_cvt_bf16<<<((size_t)DDIM * DDIM + 255) / 256, 256, 0, stream>>>(conv_w[1], w1b, (size_t)DDIM * DDIM);
    k_cvt_bf16<<<((size_t)DDIM * DDIM + 255) / 256, 256, 0, stream>>>(conv_w[2], w2b, (size_t)DDIM * DDIM);
    k_cvt_bf16<<<((size_t)NDIM * DDIM * 256 + 255) / 256, 256, 0, stream>>>(w1, wh1b, (size_t)NDIM * DDIM * 256);

    // 4) layer 0: y = sc @ W0  (flat 204800x400 @ 400x128, f32 A converted on the fly)
    k_gemm<true, true><<<dim3(MROWS / 128, 1), 256, 0, stream>>>(sc, w0b, y_bf, nullptr, MROWS, DDIM, NDIM);
    k_agg<<<dim3(BDIM, 4), 256, 0, stream>>>(a_bf, y_bf, xA, conv_b[0], bn_g[0], bn_b[0], bn_m[0], bn_v[0], 1);

    // 5) layer 1
    k_gemm<false, true><<<dim3(MROWS / 128, 1), 256, 0, stream>>>(xA, w1b, y_bf, nullptr, MROWS, DDIM, DDIM);
    k_agg<<<dim3(BDIM, 4), 256, 0, stream>>>(a_bf, y_bf, xB, conv_b[1], bn_g[1], bn_b[1], bn_m[1], bn_v[1], 1);

    // 6) layer 2 (no ReLU)
    k_gemm<false, true><<<dim3(MROWS / 128, 1), 256, 0, stream>>>(xB, w2b, y_bf, nullptr, MROWS, DDIM, DDIM);
    k_agg<<<dim3(BDIM, 4), 256, 0, stream>>>(a_bf, y_bf, xA, conv_b[2], bn_g[2], bn_b[2], bn_m[2], bn_v[2], 0);

    // 7) head fc1: [512, 51200] @ [51200, 256] + b1
    k_gemm<false, false><<<dim3(BDIM / 128, 2), 256, 0, stream>>>(xA, wh1b, h1raw, b1, BDIM, 256, NDIM * DDIM);
    k_ln_relu<<<(BDIM + 7) / 8, 256, 0, stream>>>(h1raw, h1, ln1_g, ln1_b, 256, BDIM, 1);

    // 8) fc2 + LN + ReLU
    k_fc<<<(BDIM * 64 + 255) / 256, 256, 0, stream>>>(h1, w2, b2, h2raw, BDIM, 64, 256);
    k_ln_relu<<<(BDIM + 7) / 8, 256, 0, stream>>>(h2raw, h2, ln2_g, ln2_b, 64, BDIM, 1);

    // 9) fc3 -> output [512, 2] f32
    k_fc<<<(BDIM * 2 + 255) / 256, 256, 0, stream>>>(h2, w3, b3, (float*)d_out, BDIM, 2, 64);
}
